// SoftCascadeLayer_5927054868499
// MI455X (gfx1250) — matrix operand order repaired and checked by reading, not compiled, not run on hardware
//
#include <hip/hip_runtime.h>
#include <hip/hip_bf16.h>

// ---------------------------------------------------------------------------
// SoftCascadeLayer for MI455X (gfx1250, wave32, WMMA).
//
// Input flattening assumption (setup_inputs dict order, depth-first):
//  0: hidden_states (8,4096,1024) f32      1: importance_scores (8,4096) f32
//  2: chunk_queries (8,1024)
//  3: chunk  w_in (3072,1024)  4: b_in (3072)  5: w_out (1024,1024)  6: b_out (1024)
//  7: prio   w_in              8: b_in         9: w_out             10: b_out
// 11: decode w_in             12: b_in        13: w_out             14: b_out
// 15: pn_g  16: pn_b  17: dn_g  18: dn_b
// ---------------------------------------------------------------------------

#define CB   8
#define CS   4096
#define CH   1024
#define CNH  4
#define CHD  256
#define CNC  8
#define CBS  (CB * CS)          // 32768
#define CBSH ((size_t)CBS * CH) // 33554432

typedef __attribute__((ext_vector_type(16))) __bf16 v16bf;
typedef __attribute__((ext_vector_type(8)))  float  v8f;

union FragBF { v16bf v; uint4 q[2]; };

__device__ __forceinline__ unsigned short f2bf(float f) {
    unsigned int u = __float_as_uint(f);
    unsigned int r = (u + 0x7FFFu + ((u >> 16) & 1u)) >> 16;
    return (unsigned short)r;
}

// ---------------------------------------------------------------------------
// Elementwise prep: hidden -> bf16, weighted = hidden*imp -> bf16
// ---------------------------------------------------------------------------
__global__ void k_prep(const float* __restrict__ hid, const float* __restrict__ imp,
                       unsigned short* __restrict__ hbf, unsigned short* __restrict__ wbf,
                       int n) {
    int i = blockIdx.x * 256 + threadIdx.x;
    if (i >= n) return;
    float h = hid[i];
    float w = h * imp[i >> 10];
    hbf[i] = f2bf(h);
    wbf[i] = f2bf(w);
}

__global__ void k_cvt(const float* __restrict__ src, unsigned short* __restrict__ dst, int n) {
    int i = blockIdx.x * 256 + threadIdx.x;
    if (i < n) dst[i] = f2bf(src[i]);
}

// ---------------------------------------------------------------------------
// Big GEMM: C(M,N) = A(M,K)_bf16 * W(N,K)_bf16^T + bias   (fp32 out)
// Block = 256 thr = 8 waves; block tile 128(M) x 128(N); wave tile 32 x 64.
// Per K-step (32): 2 A frags + 4 B frags (12 x b128 loads) -> 8 WMMAs
// (1.5 loads per v_wmma_f32_16x16x32_bf16).
// A frag (16x32 bf16): lane m=lane&15, kb=(lane<16)?0:16, k = kb+j
//   => two contiguous 16B loads at k0+kb and k0+kb+8.
// B frag (32x16 bf16): lane n=lane&15, kb=(lane<16)?0:16, k = kb+j
//   => two contiguous 16B loads (32B run).
// D: VGPR r, lane L -> m = r + ((L<16)?0:8), n = L&15.
// ---------------------------------------------------------------------------
__global__ __launch_bounds__(256) void k_gemm_bf16(
    const unsigned short* __restrict__ A, const unsigned short* __restrict__ W,
    const float* __restrict__ bias, float* __restrict__ C, int M, int N, int K) {
    const int lane = threadIdx.x & 31;
    const int wave = threadIdx.x >> 5;
    const int row0 = blockIdx.y * 128 + (wave & 3) * 32;
    const int col0 = blockIdx.x * 128 + (wave >> 2) * 64;

    const int kbA = (lane < 16) ? 0 : 16;
    const int nB  = lane & 15;
    const int kbB = (lane < 16) ? 0 : 16;

    const unsigned short* __restrict__ arow0 = A + (size_t)(row0 + (lane & 15)) * K;
    const unsigned short* __restrict__ arow1 = arow0 + (size_t)16 * K;
    const unsigned short* __restrict__ wrow  = W + (size_t)(col0 + nB) * K;

    v8f acc00 = {}, acc01 = {}, acc02 = {}, acc03 = {};
    v8f acc10 = {}, acc11 = {}, acc12 = {}, acc13 = {};

    for (int kk = 0; kk < K; kk += 32) {
        FragBF fa0, fa1;
        fa0.q[0] = *(const uint4*)(arow0 + kk + kbA);
        fa0.q[1] = *(const uint4*)(arow0 + kk + kbA + 8);
        fa1.q[0] = *(const uint4*)(arow1 + kk + kbA);
        fa1.q[1] = *(const uint4*)(arow1 + kk + kbA + 8);

        const unsigned short* wp = wrow + kk + kbB;
        FragBF fb0, fb1, fb2, fb3;
        fb0.q[0] = *(const uint4*)(wp);
        fb0.q[1] = *(const uint4*)(wp + 8);
        fb1.q[0] = *(const uint4*)(wp + (size_t)16 * K);
        fb1.q[1] = *(const uint4*)(wp + (size_t)16 * K + 8);
        fb2.q[0] = *(const uint4*)(wp + (size_t)32 * K);
        fb2.q[1] = *(const uint4*)(wp + (size_t)32 * K + 8);
        fb3.q[0] = *(const uint4*)(wp + (size_t)48 * K);
        fb3.q[1] = *(const uint4*)(wp + (size_t)48 * K + 8);

        acc00 = __builtin_amdgcn_wmma_f32_16x16x32_bf16(false, fa0.v, false, fb0.v, (short)0, acc00, false, false);
        acc10 = __builtin_amdgcn_wmma_f32_16x16x32_bf16(false, fa1.v, false, fb0.v, (short)0, acc10, false, false);
        acc01 = __builtin_amdgcn_wmma_f32_16x16x32_bf16(false, fa0.v, false, fb1.v, (short)0, acc01, false, false);
        acc11 = __builtin_amdgcn_wmma_f32_16x16x32_bf16(false, fa1.v, false, fb1.v, (short)0, acc11, false, false);
        acc02 = __builtin_amdgcn_wmma_f32_16x16x32_bf16(false, fa0.v, false, fb2.v, (short)0, acc02, false, false);
        acc12 = __builtin_amdgcn_wmma_f32_16x16x32_bf16(false, fa1.v, false, fb2.v, (short)0, acc12, false, false);
        acc03 = __builtin_amdgcn_wmma_f32_16x16x32_bf16(false, fa0.v, false, fb3.v, (short)0, acc03, false, false);
        acc13 = __builtin_amdgcn_wmma_f32_16x16x32_bf16(false, fa1.v, false, fb3.v, (short)0, acc13, false, false);
    }

    const int mOut0 = row0 + ((lane < 16) ? 0 : 8);
    const int mOut1 = mOut0 + 16;
    const int n0 = col0 + (lane & 15);
    const float b0 = bias[n0], b1 = bias[n0 + 16], b2 = bias[n0 + 32], b3 = bias[n0 + 48];
#pragma unroll
    for (int r = 0; r < 8; ++r) {
        size_t ro0 = (size_t)(mOut0 + r) * N;
        size_t ro1 = (size_t)(mOut1 + r) * N;
        C[ro0 + n0]      = acc00[r] + b0;
        C[ro0 + n0 + 16] = acc01[r] + b1;
        C[ro0 + n0 + 32] = acc02[r] + b2;
        C[ro0 + n0 + 48] = acc03[r] + b3;
        C[ro1 + n0]      = acc10[r] + b0;
        C[ro1 + n0 + 16] = acc11[r] + b1;
        C[ro1 + n0 + 32] = acc12[r] + b2;
        C[ro1 + n0 + 48] = acc13[r] + b3;
    }
}

// ---------------------------------------------------------------------------
// Generic row projection: Y[r,n] = dot(X[r,:K], W[n,:K]) + bias[n]
// One thread per output element (small fp32 GEMMs only).
// ---------------------------------------------------------------------------
__global__ void k_proj_rows(const float* __restrict__ X, const float* __restrict__ W,
                            const float* __restrict__ bias, float* __restrict__ Y,
                            int rows, int N, int K) {
    int idx = blockIdx.x * 256 + threadIdx.x;
    if (idx >= rows * N) return;
    int r = idx / N, n = idx - r * N;
    const float4* x = (const float4*)(X + (size_t)r * K);
    const float4* w = (const float4*)(W + (size_t)n * K);
    float a = 0.f;
    for (int k = 0; k < K / 4; ++k) {
        float4 xv = x[k], wv = w[k];
        a = fmaf(xv.x, wv.x, fmaf(xv.y, wv.y, fmaf(xv.z, wv.z, fmaf(xv.w, wv.w, a))));
    }
    Y[idx] = a + bias[n];
}

// ---------------------------------------------------------------------------
// Stage 1 attention: block = (b,h). scores(8,4096) in LDS, exact softmax,
// per-head importance mass, o(8,256).
// ---------------------------------------------------------------------------
__global__ __launch_bounds__(256) void k_stage1_attn(
    const float* __restrict__ Kh, const float* __restrict__ Vh,
    const float* __restrict__ qh, const float* __restrict__ imp,
    float* __restrict__ o_out, float* __restrict__ imp_part) {
    __shared__ float sc[CNC][CS];
    __shared__ float qsh[CNC][CHD];
    __shared__ float red[256];
    const int b = blockIdx.x >> 2, h = blockIdx.x & 3;
    const int tid = threadIdx.x;

    for (int i = tid; i < CNC * CHD; i += 256)
        qsh[i >> 8][i & 255] = qh[(i >> 8) * CH + h * CHD + (i & 255)];
    __syncthreads();

    const float scale = 0.0625f; // 1/sqrt(256)
    for (int s = tid; s < CS; s += 256) {
        const float4* kr = (const float4*)(Kh + ((size_t)(b * CS + s)) * CH + h * CHD);
        float acc[CNC];
#pragma unroll
        for (int c = 0; c < CNC; ++c) acc[c] = 0.f;
        for (int k = 0; k < CHD / 4; ++k) {
            float4 kv = kr[k];
#pragma unroll
            for (int c = 0; c < CNC; ++c)
                acc[c] = fmaf(kv.x, qsh[c][4 * k], fmaf(kv.y, qsh[c][4 * k + 1],
                         fmaf(kv.z, qsh[c][4 * k + 2], fmaf(kv.w, qsh[c][4 * k + 3], acc[c]))));
        }
#pragma unroll
        for (int c = 0; c < CNC; ++c) sc[c][s] = acc[c] * scale;
    }
    __syncthreads();

    const float* impb = imp + b * CS;
    for (int c = 0; c < CNC; ++c) {
        float m = -3.0e38f;
        for (int s = tid; s < CS; s += 256) m = fmaxf(m, sc[c][s]);
        red[tid] = m; __syncthreads();
        for (int st = 128; st > 0; st >>= 1) { if (tid < st) red[tid] = fmaxf(red[tid], red[tid + st]); __syncthreads(); }
        m = red[0]; __syncthreads();

        float e = 0.f, ip = 0.f;
        for (int s = tid; s < CS; s += 256) {
            float ex = __expf(sc[c][s] - m);
            sc[c][s] = ex; e += ex; ip += ex * impb[s];
        }
        red[tid] = e; __syncthreads();
        for (int st = 128; st > 0; st >>= 1) { if (tid < st) red[tid] += red[tid + st]; __syncthreads(); }
        e = red[0]; __syncthreads();
        red[tid] = ip; __syncthreads();
        for (int st = 128; st > 0; st >>= 1) { if (tid < st) red[tid] += red[tid + st]; __syncthreads(); }
        ip = red[0]; __syncthreads();

        float inv = 1.f / e;
        for (int s = tid; s < CS; s += 256) sc[c][s] *= inv;
        if (tid == 0) imp_part[(b * CNH + h) * CNC + c] = ip * inv;
        __syncthreads();
    }

    // o[c][d] = sum_s probs[c][s] * V[s][d]   (d = tid)
    const int d = tid;
    float oacc[CNC];
#pragma unroll
    for (int c = 0; c < CNC; ++c) oacc[c] = 0.f;
    const float* vp = Vh + (size_t)b * CS * CH + h * CHD + d;
    for (int s = 0; s < CS; ++s) {
        float vv = vp[(size_t)s * CH];
#pragma unroll
        for (int c = 0; c < CNC; ++c) oacc[c] = fmaf(sc[c][s], vv, oacc[c]);
    }
#pragma unroll
    for (int c = 0; c < CNC; ++c)
        o_out[((size_t)(b * CNC + c)) * CH + h * CHD + d] = oacc[c];
}

// ---------------------------------------------------------------------------
// Per-batch: chunk_imp = mean_h(imp_part), stable descending argsort of 8,
// gather sorted_chunks; outrow[b*8+t] = b*8 + sort_idx[t] (for un-sort write).
// ---------------------------------------------------------------------------
__global__ void k_sort_gather(const float* __restrict__ imp_part,
                              const float* __restrict__ chunk_repr,
                              float* __restrict__ sorted, int* __restrict__ outrow) {
    const int b = blockIdx.x, tid = threadIdx.x;
    __shared__ int sidx[CNC];
    if (tid == 0) {
        float ci[CNC]; int idx[CNC];
        for (int c = 0; c < CNC; ++c) {
            float s = 0.f;
            for (int h = 0; h < CNH; ++h) s += imp_part[(b * CNH + h) * CNC + c];
            ci[c] = s * 0.25f; idx[c] = c;
        }
        for (int i = 1; i < CNC; ++i) {       // stable insertion, descending
            int j = i;
            while (j > 0 && ci[idx[j]] > ci[idx[j - 1]]) { int t = idx[j]; idx[j] = idx[j - 1]; idx[j - 1] = t; --j; }
        }
        for (int t = 0; t < CNC; ++t) { sidx[t] = idx[t]; outrow[b * CNC + t] = b * CNC + idx[t]; }
    }
    __syncthreads();
    for (int i = tid; i < CNC * CH; i += 256)
        sorted[(size_t)b * CNC * CH + i] =
            chunk_repr[(size_t)b * CNC * CH + sidx[i >> 10] * CH + (i & 1023)];
}

// ---------------------------------------------------------------------------
// Stage 2 causal attention over 8 chunks. qkv layout (b, t, 3072).
// ---------------------------------------------------------------------------
__global__ void k_s2_attn(const float* __restrict__ qkv, float* __restrict__ o) {
    const int b = blockIdx.x, tid = threadIdx.x;
    __shared__ float sp[CNH][CNC][CNC];
    const float* base = qkv + (size_t)b * CNC * 3072;
    {
        int h = tid >> 6, t = (tid >> 3) & 7, u = tid & 7;
        const float* qp = base + t * 3072 + h * CHD;
        const float* kp = base + u * 3072 + 1024 + h * CHD;
        float a = 0.f;
        for (int d = 0; d < CHD; ++d) a = fmaf(qp[d], kp[d], a);
        sp[h][t][u] = a * 0.0625f;
    }
    __syncthreads();
    if (tid < 32) {
        int h = tid >> 3, t = tid & 7;
        float m = -3.0e38f;
        for (int u = 0; u <= t; ++u) m = fmaxf(m, sp[h][t][u]);
        float tmp[CNC]; float e = 0.f;
        for (int u = 0; u < CNC; ++u) {
            float v = (u <= t) ? __expf(sp[h][t][u] - m) : 0.f;
            tmp[u] = v; e += v;
        }
        float inv = 1.f / e;
        for (int u = 0; u < CNC; ++u) sp[h][t][u] = tmp[u] * inv;
    }
    __syncthreads();
    for (int i = tid; i < CNC * CH; i += 256) {
        int t = i >> 10, n = i & 1023, h = n >> 8;
        float a = 0.f;
#pragma unroll
        for (int u = 0; u < CNC; ++u) a = fmaf(sp[h][t][u], base[u * 3072 + 2048 + n], a);
        o[(size_t)b * CNC * CH + i] = a;
    }
}

// ---------------------------------------------------------------------------
// add + layernorm; optional row remap on the OUTPUT (for un-sort scatter).
// One block per row of 1024.
// ---------------------------------------------------------------------------
__global__ __launch_bounds__(256) void k_add_ln(
    const float* __restrict__ x1, const float* __restrict__ x2,
    const float* __restrict__ g, const float* __restrict__ bt,
    float* __restrict__ out, const int* __restrict__ row_map) {
    __shared__ float red[256];
    const int r = blockIdx.x, tid = threadIdx.x;
    const float* a = x1 + (size_t)r * CH;
    const float* c = x2 + (size_t)r * CH;
    float lv[4]; float s = 0.f;
#pragma unroll
    for (int i = 0; i < 4; ++i) { lv[i] = a[tid + i * 256] + c[tid + i * 256]; s += lv[i]; }
    red[tid] = s; __syncthreads();
    for (int st = 128; st > 0; st >>= 1) { if (tid < st) red[tid] += red[tid + st]; __syncthreads(); }
    float mean = red[0] * (1.f / CH); __syncthreads();
    float vs = 0.f;
#pragma unroll
    for (int i = 0; i < 4; ++i) { float d = lv[i] - mean; vs += d * d; }
    red[tid] = vs; __syncthreads();
    for (int st = 128; st > 0; st >>= 1) { if (tid < st) red[tid] += red[tid + st]; __syncthreads(); }
    float inv = rsqrtf(red[0] * (1.f / CH) + 1e-5f);
    size_t orow = row_map ? (size_t)row_map[r] : (size_t)r;
#pragma unroll
    for (int i = 0; i < 4; ++i) {
        int col = tid + i * 256;
        out[orow * CH + col] = (lv[i] - mean) * inv * g[col] + bt[col];
    }
}

// ---------------------------------------------------------------------------
// Decode attention: every token attends over 8 chunks (per head).
// Block handles (b, 64 tokens); K/V for all heads staged in LDS.
// Output written directly as bf16 for the output-projection WMMA GEMM.
// ---------------------------------------------------------------------------
__global__ __launch_bounds__(256) void k_decode_attn(
    const float* __restrict__ Qd, const float* __restrict__ kv,
    unsigned short* __restrict__ out_bf) {
    __shared__ float ks[CNH][CNC][CHD];
    __shared__ float vs[CNH][CNC][CHD];
    const int b = blockIdx.y, s0 = blockIdx.x * 64, tid = threadIdx.x;
    for (int i = tid; i < CNC * CH; i += 256) {
        int t = i >> 10, n = i & 1023;
        ks[n >> 8][t][n & 255] = kv[(size_t)(b * CNC + t) * 2048 + n];
        vs[n >> 8][t][n & 255] = kv[(size_t)(b * CNC + t) * 2048 + 1024 + n];
    }
    __syncthreads();
    const int tok = s0 + (tid >> 2), h = tid & 3;
    const float4* q = (const float4*)(Qd + ((size_t)(b * CS + tok)) * CH + h * CHD);
    float sc8[CNC];
#pragma unroll
    for (int j = 0; j < CNC; ++j) sc8[j] = 0.f;
    for (int k = 0; k < CHD / 4; ++k) {
        float4 qv = q[k];
#pragma unroll
        for (int j = 0; j < CNC; ++j)
            sc8[j] = fmaf(qv.x, ks[h][j][4 * k], fmaf(qv.y, ks[h][j][4 * k + 1],
                     fmaf(qv.z, ks[h][j][4 * k + 2], fmaf(qv.w, ks[h][j][4 * k + 3], sc8[j]))));
    }
    float m = -3.0e38f;
#pragma unroll
    for (int j = 0; j < CNC; ++j) { sc8[j] *= 0.0625f; m = fmaxf(m, sc8[j]); }
    float e = 0.f;
#pragma unroll
    for (int j = 0; j < CNC; ++j) { sc8[j] = __expf(sc8[j] - m); e += sc8[j]; }
    float inv = 1.f / e;
#pragma unroll
    for (int j = 0; j < CNC; ++j) sc8[j] *= inv;

    unsigned short* op = out_bf + ((size_t)(b * CS + tok)) * CH + h * CHD;
    for (int d = 0; d < CHD; ++d) {
        float a = 0.f;
#pragma unroll
        for (int j = 0; j < CNC; ++j) a = fmaf(sc8[j], vs[h][j][d], a);
        op[d] = f2bf(a);
    }
}

// ---------------------------------------------------------------------------
extern "C" void kernel_launch(void* const* d_in, const int* in_sizes, int n_in,
                              void* d_out, int out_size, void* d_ws, size_t ws_size,
                              hipStream_t stream) {
    const float* hidden  = (const float*)d_in[0];
    const float* imp     = (const float*)d_in[1];
    const float* cq      = (const float*)d_in[2];
    const float* ca_win  = (const float*)d_in[3];
    const float* ca_bin  = (const float*)d_in[4];
    const float* ca_wout = (const float*)d_in[5];
    const float* ca_bout = (const float*)d_in[6];
    const float* pa_win  = (const float*)d_in[7];
    const float* pa_bin  = (const float*)d_in[8];
    const float* pa_wout = (const float*)d_in[9];
    const float* pa_bout = (const float*)d_in[10];
    const float* da_win  = (const float*)d_in[11];
    const float* da_bin  = (const float*)d_in[12];
    const float* da_wout = (const float*)d_in[13];
    const float* da_bout = (const float*)d_in[14];
    const float* pn_g    = (const float*)d_in[15];
    const float* pn_b    = (const float*)d_in[16];
    const float* dn_g    = (const float*)d_in[17];
    const float* dn_b    = (const float*)d_in[18];
    float* out = (float*)d_out;

    char* ws = (char*)d_ws;
    size_t off = 0;
    auto take = [&](size_t bytes) { char* p = ws + off; off += (bytes + 255) & ~(size_t)255; return p; };

    unsigned short* hbf    = (unsigned short*)take(CBSH * 2);      // hidden bf16
    unsigned short* wbf    = (unsigned short*)take(CBSH * 2);      // weighted bf16 -> reused as decode-attn bf16
    float*          bigC   = (float*)take(CBSH * 4);               // Kh -> Qd
    float*          bigD   = (float*)take(CBSH * 4);               // Vh -> dec_proj
    unsigned short* w_kc   = (unsigned short*)take(1024 * 1024 * 2);
    unsigned short* w_vc   = (unsigned short*)take(1024 * 1024 * 2);
    unsigned short* w_qd   = (unsigned short*)take(1024 * 1024 * 2);
    unsigned short* w_od   = (unsigned short*)take(1024 * 1024 * 2);
    float* qh1    = (float*)take(CNC * CH * 4);                    // stage1 projected queries (8,1024)
    float* o1     = (float*)take(CB * CNC * CH * 4);               // stage1 attn out (b,c,1024)
    float* imppt  = (float*)take(CB * CNH * CNC * 4);
    float* crep   = (float*)take(CB * CNC * CH * 4);               // chunk_repr
    int*   outrow = (int*)take(CB * CNC * 4);
    float* sorted = (float*)take(CB * CNC * CH * 4);
    float* s2qkv  = (float*)take(CB * CNC * 3072 * 4);
    float* s2ao   = (float*)take(CB * CNC * CH * 4);
    float* s2pr   = (float*)take(CB * CNC * CH * 4);
    float* pch    = (float*)take(CB * CNC * CH * 4);               // pchunks
    float* dkv    = (float*)take(CB * CNC * 2048 * 4);             // decode K|V per chunk
    (void)ws_size; (void)in_sizes; (void)n_in; (void)out_size;

    const int n = (int)CBSH;
    // 1) prep activations + weight conversion
    k_prep<<<dim3(n / 256), dim3(256), 0, stream>>>(hidden, imp, hbf, wbf, n);
    k_cvt<<<dim3(4096), dim3(256), 0, stream>>>(ca_win + 1024 * 1024, w_kc, 1024 * 1024);
    k_cvt<<<dim3(4096), dim3(256), 0, stream>>>(ca_win + 2048 * 1024, w_vc, 1024 * 1024);
    k_cvt<<<dim3(4096), dim3(256), 0, stream>>>(da_win,               w_qd, 1024 * 1024);
    k_cvt<<<dim3(4096), dim3(256), 0, stream>>>(da_wout,              w_od, 1024 * 1024);

    // 2) Stage-1 K/V projections (big WMMA GEMMs): (32768,1024) = A(32768,1024)*W^T
    dim3 ggrid(CH / 128, CBS / 128), gblk(256);
    k_gemm_bf16<<<ggrid, gblk, 0, stream>>>(wbf, w_kc, ca_bin + 1024, bigC, CBS, CH, CH); // Kh
    k_gemm_bf16<<<ggrid, gblk, 0, stream>>>(hbf, w_vc, ca_bin + 2048, bigD, CBS, CH, CH); // Vh

    // 3) Stage-1 query projection (tiny): qh(8,1024) = chunk_queries @ wq^T + bq
    k_proj_rows<<<dim3(32), dim3(256), 0, stream>>>(cq, ca_win, ca_bin, qh1, CNC, CH, CH);

    // 4) Stage-1 attention per (b,h): softmax over S, o, importance mass
    k_stage1_attn<<<dim3(CB * CNH), dim3(256), 0, stream>>>(bigC, bigD, qh1, imp, o1, imppt);

    // 5) Stage-1 out projection -> chunk_repr (64 rows)
    k_proj_rows<<<dim3(256), dim3(256), 0, stream>>>(o1, ca_wout, ca_bout, crep, CB * CNC, CH, CH);

    // 6) sort by chunk importance, gather
    k_sort_gather<<<dim3(CB), dim3(256), 0, stream>>>(imppt, crep, sorted, outrow);

    // 7) Stage-2 QKV projection (64 rows x 3072)
    k_proj_rows<<<dim3(768), dim3(256), 0, stream>>>(sorted, pa_win, pa_bin, s2qkv, CB * CNC, 3072, CH);

    // 8) Stage-2 causal attention
    k_s2_attn<<<dim3(CB), dim3(256), 0, stream>>>(s2qkv, s2ao);

    // 9) Stage-2 out projection, then residual + LN with un-sort scatter -> pchunks
    k_proj_rows<<<dim3(256), dim3(256), 0, stream>>>(s2ao, pa_wout, pa_bout, s2pr, CB * CNC, CH, CH);
    k_add_ln<<<dim3(CB * CNC), dim3(256), 0, stream>>>(s2pr, sorted, pn_g, pn_b, pch, outrow);

    // 10) Decode K/V projection from pchunks (64 rows x 2048)
    k_proj_rows<<<dim3(512), dim3(256), 0, stream>>>(pch, da_win + 1024 * 1024, da_bin + 1024, dkv, CB * CNC, 2048, CH);

    // 11) Decode Q projection (big WMMA GEMM) -> bigC (Kh no longer needed)
    k_gemm_bf16<<<ggrid, gblk, 0, stream>>>(hbf, w_qd, da_bin, bigC, CBS, CH, CH);

    // 12) Decode attention; bf16 output into wbf (weighted no longer needed)
    k_decode_attn<<<dim3(CS / 64, CB), dim3(256), 0, stream>>>(bigC, dkv, wbf);

    // 13) Decode out projection (big WMMA GEMM) -> bigD (Vh no longer needed)
    k_gemm_bf16<<<ggrid, gblk, 0, stream>>>(wbf, w_od, da_bout, bigD, CBS, CH, CH);

    // 14) final residual + layernorm -> d_out
    k_add_ln<<<dim3(CBS), dim3(256), 0, stream>>>(bigD, hidden, dn_g, dn_b, out, nullptr);
}
